// CRF_18571438588055
// MI455X (gfx1250) — matrix operand order repaired and checked by reading, not compiled, not run on hardware
//
#include <hip/hip_runtime.h>
#include <hip/hip_bf16.h>

#define CRF_BS   512
#define CRF_SEQ  768
#define CRF_NL   32

typedef __attribute__((ext_vector_type(16))) _Float16 v16h;
typedef __attribute__((ext_vector_type(2)))  _Float16 h2;
typedef __attribute__((ext_vector_type(8)))  float    v8f;

// A-matrix (16x32 f16) slot->K map: lanes 0-15 hold K{0..7,16..23}, lanes 16-31 +8.
__device__ __forceinline__ int kA_of(int v, int hiA) {
    return (v < 8 ? v : v + 8) + hiA;   // hiA = 0 or 8
}

// All-reduce max within each 16-lane DPP row (wave32 => two independent rows).
// Stages: xor1 (quad_perm[1,0,3,2]), xor2 (quad_perm[2,3,0,1]),
//         row_half_mirror (combines quads within 8), row_mirror (combines 8-halves).
__device__ __forceinline__ float rowmax16(float x) {
    int t;
    t = __builtin_amdgcn_update_dpp(__builtin_bit_cast(int, x), __builtin_bit_cast(int, x),
                                    0xB1, 0xF, 0xF, true);   // quad_perm [1,0,3,2]
    x = fmaxf(x, __builtin_bit_cast(float, t));
    t = __builtin_amdgcn_update_dpp(__builtin_bit_cast(int, x), __builtin_bit_cast(int, x),
                                    0x4E, 0xF, 0xF, true);   // quad_perm [2,3,0,1]
    x = fmaxf(x, __builtin_bit_cast(float, t));
    t = __builtin_amdgcn_update_dpp(__builtin_bit_cast(int, x), __builtin_bit_cast(int, x),
                                    0x141, 0xF, 0xF, true);  // row_half_mirror
    x = fmaxf(x, __builtin_bit_cast(float, t));
    t = __builtin_amdgcn_update_dpp(__builtin_bit_cast(int, x), __builtin_bit_cast(int, x),
                                    0x140, 0xF, 0xF, true);  // row_mirror
    x = fmaxf(x, __builtin_bit_cast(float, t));
    return x;
}

__global__ __launch_bounds__(32) void crf_fwd_kernel(
    const float* __restrict__ X, const int* __restrict__ Y,
    const unsigned char* __restrict__ mask,
    const float* __restrict__ Tr, const float* __restrict__ st,
    const float* __restrict__ en, float* __restrict__ out)
{
    __shared__ _Float16 uT[16][34];   // u (normalized exp-alpha), row-major, padded
    __shared__ float    sRow[16];     // per-row log-scale at the end

    const int lane = threadIdx.x;        // 0..31, wave32
    const int b0   = blockIdx.x * 16;    // batch tile base
    const int cl   = lane & 15;          // column within a 16-wide N tile
    const int hiA  = (lane >> 4) ? 8 : 0;   // A K-offset; also D-layout row offset
    const int hiB  = (lane >> 4) ? 16 : 0;  // B K-offset

    // ---- Loop-invariant B operands: E = exp(transition), WMMA B layout ----
    v16h B0, B1;
#pragma unroll
    for (int v = 0; v < 16; ++v) {
        int k = kA_of(v, hiA);
        B0[v] = (_Float16)__expf(Tr[k * CRF_NL + cl]);
        B1[v] = (_Float16)__expf(Tr[k * CRF_NL + cl + 16]);
    }

    // ---- t = 0: alpha0 = start + X[:,0,:] in D layout; normalize into LDS ----
    float sAcc[8];
    {
        const float st0 = st[cl];
        const float st1 = st[cl + 16];
#pragma unroll
        for (int r = 0; r < 8; ++r) {
            const float* xp = X + ((size_t)(b0 + r + hiA) * CRF_SEQ) * CRF_NL;
            float a0 = st0 + xp[cl];
            float a1 = st1 + xp[cl + 16];
            float m = rowmax16(fmaxf(a0, a1));
            sAcc[r] = m;
            uT[r + hiA][cl]      = (_Float16)__expf(a0 - m);
            uT[r + hiA][cl + 16] = (_Float16)__expf(a1 - m);
        }
    }
    __syncthreads();

    // ---- preload x for t=1 ----
    float xr0[8], xr1[8];
#pragma unroll
    for (int r = 0; r < 8; ++r) {
        const float* xp = X + ((size_t)(b0 + r + hiA) * CRF_SEQ + 1) * CRF_NL;
        xr0[r] = xp[cl];
        xr1[r] = xp[cl + 16];
    }

    // ---- sequential scan: q = u * E (WMMA), w = q * exp(x), renormalize ----
    for (int t = 1; t < CRF_SEQ; ++t) {
        // stream prefetch ~24 steps ahead; lanes 0-15 cover all 16 rows' lines
        {
            const int tf = (t + 24 < CRF_SEQ) ? t + 24 : CRF_SEQ - 1;
            const float* pf = X + ((size_t)(b0 + cl) * CRF_SEQ + tf) * CRF_NL;
            __builtin_prefetch(pf, 0, 0);
        }

        // exp of current emissions (TRANS ops co-execute with WMMA)
        float ex0[8], ex1[8];
#pragma unroll
        for (int r = 0; r < 8; ++r) { ex0[r] = __expf(xr0[r]); ex1[r] = __expf(xr1[r]); }

        // gather A operand (u) from LDS in WMMA A layout; pairs as b32 reads
        v16h A;
#pragma unroll
        for (int v = 0; v < 16; v += 2) {
            int k = kA_of(v, hiA);               // even -> 4B aligned (row stride 68B)
            h2 pr = *(const h2*)&uT[cl][k];
            A[v] = pr.x; A[v + 1] = pr.y;
        }

        // prefetch x for t+1 (clamped; overlaps with WMMA + VALU below)
        float xn0[8], xn1[8];
        {
            const int tp = (t + 1 < CRF_SEQ) ? (t + 1) : t;
#pragma unroll
            for (int r = 0; r < 8; ++r) {
                const float* xp = X + ((size_t)(b0 + r + hiA) * CRF_SEQ + tp) * CRF_NL;
                xn0[r] = xp[cl];
                xn1[r] = xp[cl + 16];
            }
        }

        v8f C0 = {0.f,0.f,0.f,0.f,0.f,0.f,0.f,0.f};
        v8f C1 = {0.f,0.f,0.f,0.f,0.f,0.f,0.f,0.f};
        C0 = __builtin_amdgcn_wmma_f32_16x16x32_f16(false, A, false, B0, (short)0, C0, false, false);
        C1 = __builtin_amdgcn_wmma_f32_16x16x32_f16(false, A, false, B1, (short)0, C1, false, false);

        // w = q * exp(x); per-row max via DPP16 all-reduce (no LDS on critical path)
        float w0[8], w1[8], mx[8];
#pragma unroll
        for (int r = 0; r < 8; ++r) {
            w0[r] = C0[r] * ex0[r];
            w1[r] = C1[r] * ex1[r];
            mx[r] = rowmax16(fmaxf(w0[r], w1[r]));
        }

        // masked update: u' = w * rcp(max) into LDS, s += log(max); mask==0 keeps old row
#pragma unroll
        for (int r = 0; r < 8; ++r) {
            unsigned char mk = mask[(size_t)(b0 + r + hiA) * CRF_SEQ + t];
            if (mk) {
                float inv = __builtin_amdgcn_rcpf(mx[r]);
                uT[r + hiA][cl]      = (_Float16)(w0[r] * inv);
                uT[r + hiA][cl + 16] = (_Float16)(w1[r] * inv);
                sAcc[r] += __logf(mx[r]);
            }
        }
        __syncthreads();

#pragma unroll
        for (int r = 0; r < 8; ++r) { xr0[r] = xn0[r]; xr1[r] = xn1[r]; }
    }

    // ---- finalize log_Z: s + log(sum_j u_j * exp(end_j)) ----
    if (lane == 0) {
#pragma unroll
        for (int r = 0; r < 8; ++r) sRow[r] = sAcc[r];
    }
    if (lane == 16) {
#pragma unroll
        for (int r = 0; r < 8; ++r) sRow[r + 8] = sAcc[r];
    }
    __syncthreads();

    float logZ = 0.0f;
    if (lane < 16) {
        float acc = 0.0f;
#pragma unroll
        for (int j = 0; j < CRF_NL; ++j)
            acc += (float)uT[lane][j] * __expf(en[j]);
        logZ = sRow[lane] + __logf(acc);
    }

    // ---- numerator: gold-path score; halves split the time range ----
    const int row  = cl;
    const int half = lane >> 4;
    const int b    = b0 + row;
    const int*           Yb = Y    + (size_t)b * CRF_SEQ;
    const unsigned char* Mb = mask + (size_t)b * CRF_SEQ;
    const float*         Xb = X    + (size_t)b * CRF_SEQ * CRF_NL;

    const int t0 = half ? 384 : 1;
    const int t1 = half ? CRF_SEQ : 384;

    float sumN = 0.0f;
    int   cnt  = 0;
    int yprev = Yb[t0 - 1];
    for (int t = t0; t < t1; ++t) {
        int yt = Yb[t];
        int mb = Mb[t];
        float mf = (float)mb;
        sumN += (Tr[yprev * CRF_NL + yt] + Xb[(size_t)t * CRF_NL + yt]) * mf;
        cnt  += mb;
        yprev = yt;
    }
    if (half == 0) cnt += Mb[0];           // t=0 counts toward each_len
    sumN += __shfl_xor(sumN, 16, 32);
    cnt  += __shfl_xor(cnt, 16, 32);

    if (lane < 16) {
        int   y0    = Yb[0];
        float score = st[y0] + Xb[y0] + sumN;
        int lastT = cnt - 1;
        if (lastT < 0) lastT = 0;
        int ylast = Yb[lastT];
        score += en[ylast] * (float)Mb[lastT];
        out[b] = score - logZ;
    }
}

extern "C" void kernel_launch(void* const* d_in, const int* in_sizes, int n_in,
                              void* d_out, int out_size, void* d_ws, size_t ws_size,
                              hipStream_t stream) {
    (void)in_sizes; (void)n_in; (void)d_ws; (void)ws_size; (void)out_size;
    const float*         X    = (const float*)d_in[0];
    const int*           Y    = (const int*)d_in[1];
    const unsigned char* mask = (const unsigned char*)d_in[2];
    const float*         Tr   = (const float*)d_in[3];
    const float*         st   = (const float*)d_in[4];
    const float*         en   = (const float*)d_in[5];
    float*               out  = (float*)d_out;

    dim3 grid(CRF_BS / 16);   // 32 batch tiles, one wave32 each
    dim3 block(32);
    hipLaunchKernelGGL(crf_fwd_kernel, grid, block, 0, stream,
                       X, Y, mask, Tr, st, en, out);
}
